// LlamaAttention_71081708748881
// MI455X (gfx1250) — matrix operand order repaired and checked by reading, not compiled, not run on hardware
//
#include <hip/hip_runtime.h>
#include <cstdint>
#include <cstddef>

// ---------------------------------------------------------------------------
// MX-quantized Llama attention for MI455X (gfx1250, wave32, WMMA).
// All matmuls run through v_wmma_f32_16x16x32_bf16: MX fake-quant values
// (<=4 significand bits, f32 exponent range) are exactly representable in
// bf16, so the bf16 WMMA path reproduces the reference matmul inputs exactly.
// ---------------------------------------------------------------------------

typedef __bf16 bf16;
typedef __attribute__((ext_vector_type(16))) __bf16   v16bf;
typedef __attribute__((ext_vector_type(8)))  float    v8f;
typedef __attribute__((ext_vector_type(4)))  unsigned uintx4;

union Frag { v16bf v; uintx4 q[2]; };

__device__ inline v8f v8f_zero() {
  v8f z;
  for (int i = 0; i < 8; ++i) z[i] = 0.0f;
  return z;
}

// ---------------------------------------------------------------------------
// MX block quantization: one wave32 handles one block of 32 (lane/element).
// shared_exp = floor(log2(amax)); scale = 2^(shared_exp-2); quantize with
// mbits mantissa bits, clamp to +-maxval, RNE rounding (rintf == jnp.round).
// ---------------------------------------------------------------------------
__global__ __launch_bounds__(256) void mx_quant_kernel(
    const float* __restrict__ x, bf16* __restrict__ y,
    long long n, float mbits, float maxval)
{
  long long gid = (long long)blockIdx.x * blockDim.x + threadIdx.x;
  if (gid >= n) return;
  float v = x[gid];
  float a = fabsf(v);
#pragma unroll
  for (int m = 16; m >= 1; m >>= 1) a = fmaxf(a, __shfl_xor(a, m, 32));
  float res = 0.0f;
  if (a > 0.0f) {
    float scale = exp2f(floorf(log2f(a)) - 2.0f);       // EMAX = 2
    float yv = v / scale;
    float ay = fabsf(yv);
    float e  = fminf(fmaxf(floorf(log2f(ay > 0.0f ? ay : 1.0f)), 0.0f), 2.0f);
    float lsb = exp2f(e - mbits);
    float q = rintf(yv / lsb) * lsb;
    q = fminf(fmaxf(q, -maxval), maxval);
    res = q * scale;
  }
  y[gid] = (bf16)res;
}

// ---------------------------------------------------------------------------
// C(MxN) = A(MxK) * W(NxK)^T, bf16 in / f32 out, WMMA 16x16x32.
// Wave tile 64x64 (4x4 WMMA tiles), 8 waves (2x4) -> block tile 128x256.
// A-frag per lane: row m0+mt*16+(lane&15), chunks at k+(lane>>4)*8 and +16.
// B-frag per lane: W row n0+nt*16+(lane&15), chunks at k+(lane>>4)*8 and +16.
// Weights in bf16 total ~84MB -> resident in 192MB L2 across the GEMMs.
// ---------------------------------------------------------------------------
__global__ __launch_bounds__(256) void gemm_bf16_nt(
    const bf16* __restrict__ A, const bf16* __restrict__ W,
    float* __restrict__ C, int M, int N, int K)
{
  const int wid  = threadIdx.x >> 5;
  const int lane = threadIdx.x & 31;
  const int lh = lane >> 4, ll = lane & 15;
  const int m0 = blockIdx.y * 128 + (wid & 1) * 64;
  const int n0 = blockIdx.x * 256 + (wid >> 1) * 64;
  (void)M;

  v8f acc[4][4];
#pragma unroll
  for (int i = 0; i < 4; ++i)
#pragma unroll
    for (int j = 0; j < 4; ++j) acc[i][j] = v8f_zero();

  const bf16* arow[4];
  const bf16* brow[4];
#pragma unroll
  for (int t = 0; t < 4; ++t) {
    arow[t] = A + (size_t)(m0 + t * 16 + ll) * K + lh * 8;
    brow[t] = W + (size_t)(n0 + t * 16 + ll) * K + lh * 8;
  }

  for (int k = 0; k < K; k += 32) {
    Frag a[4], b[4];
#pragma unroll
    for (int t = 0; t < 4; ++t) {
      a[t].q[0] = *(const uintx4*)(arow[t] + k);
      a[t].q[1] = *(const uintx4*)(arow[t] + k + 16);
      b[t].q[0] = *(const uintx4*)(brow[t] + k);
      b[t].q[1] = *(const uintx4*)(brow[t] + k + 16);
    }
#pragma unroll
    for (int mt = 0; mt < 4; ++mt)
#pragma unroll
      for (int nt = 0; nt < 4; ++nt)
        acc[mt][nt] = __builtin_amdgcn_wmma_f32_16x16x32_bf16(
            false, a[mt].v, false, b[nt].v, (short)0, acc[mt][nt], false, false);
  }

#pragma unroll
  for (int mt = 0; mt < 4; ++mt)
#pragma unroll
    for (int nt = 0; nt < 4; ++nt)
#pragma unroll
      for (int r = 0; r < 8; ++r)
        C[(size_t)(m0 + mt * 16 + r + 8 * lh) * N + (n0 + nt * 16 + ll)] =
            acc[mt][nt][r];
}

// ---------------------------------------------------------------------------
// RoPE + repack f32 (B*S, nheads*D) -> bf16 (B, nheads, S, D).
// ---------------------------------------------------------------------------
__global__ __launch_bounds__(256) void rope_pack_kernel(
    const float* __restrict__ X, const float* __restrict__ cosb,
    const float* __restrict__ sinb, bf16* __restrict__ out, int nheads)
{
  int idx = blockIdx.x * blockDim.x + threadIdx.x;
  int d = idx & 127;
  int s = (idx >> 7) & 1023;
  int h = (idx >> 17) & (nheads - 1);
  int b = idx / (nheads * 1024 * 128);
  const float* row = X + (size_t)(b * 1024 + s) * (nheads * 128) + h * 128;
  float v  = row[d];
  float vr = (d < 64) ? -row[d + 64] : row[d - 64];
  size_t cs = (size_t)(b * 1024 + s) * 128 + d;
  out[idx] = (bf16)(v * cosb[cs] + vr * sinb[cs]);
}

// V: f32 (B*S, KVH*D) -> bf16 transposed (B, KVH, D, S) so the P@V
// B-fragment is two 16B loads along S per lane.
__global__ __launch_bounds__(256) void v_transpose_kernel(
    const float* __restrict__ Vf, bf16* __restrict__ Vt)
{
  int idx = blockIdx.x * blockDim.x + threadIdx.x;  // ((b*8+h)*128+d)*1024+s
  int s = idx & 1023;
  int d = (idx >> 10) & 127;
  int h = (idx >> 17) & 7;
  int b = idx >> 20;
  Vt[idx] = (bf16)Vf[(size_t)(b * 1024 + s) * 1024 + h * 128 + d];
}

// ---------------------------------------------------------------------------
// Flash-style causal attention, one wave per 16-row Q tile.
// Scores: 4 WMMAs over D=128; softmax stats via width-16 shuffles (each
// C-layout row lives in one 16-lane half); P bounced through per-wave LDS
// tile to re-layout C->A fragment; P@V: 8 WMMAs per 32-wide KV block.
// ---------------------------------------------------------------------------
__global__ __launch_bounds__(256) void attn_fwd_kernel(
    const bf16* __restrict__ Qr, const bf16* __restrict__ Kr,
    const bf16* __restrict__ Vt, float* __restrict__ AO)
{
  __shared__ __align__(16) bf16 pbuf[8 * 512];  // 16x32 bf16 per wave
  const int wid = threadIdx.x >> 5, lane = threadIdx.x & 31;
  const int lh = lane >> 4, ll = lane & 15;
  const int gw = blockIdx.x * 8 + wid;  // ((b*32+h)*64+qt)
  const int qt = gw & 63;
  const int h  = (gw >> 6) & 31;
  const int b  = gw >> 11;
  const int kvh = h >> 2;               // NH/KVH = 4
  const float scaling = 0.08838834764831845f;  // 1/sqrt(128)

  const bf16* Qbase = Qr + ((size_t)(b * 32 + h) * 1024 + qt * 16) * 128;
  const bf16* Kbase = Kr + (size_t)(b * 8 + kvh) * 1024 * 128;
  const bf16* Vbase = Vt + (size_t)(b * 8 + kvh) * 128 * 1024;

  Frag qf[4];
  {
    const bf16* qrow = Qbase + (size_t)ll * 128 + lh * 8;
#pragma unroll
    for (int kk = 0; kk < 4; ++kk) {
      qf[kk].q[0] = *(const uintx4*)(qrow + kk * 32);
      qf[kk].q[1] = *(const uintx4*)(qrow + kk * 32 + 16);
    }
  }

  v8f acc[8];
#pragma unroll
  for (int t = 0; t < 8; ++t) acc[t] = v8f_zero();
  float mrow[8], lrow[8];
#pragma unroll
  for (int r = 0; r < 8; ++r) { mrow[r] = -3.0e38f; lrow[r] = 0.0f; }

  bf16* myp = pbuf + wid * 512;
  const int nblk = (qt * 16 + 47) >> 5;  // causal: KV blocks of 32 needed

  for (int j = 0; j < nblk; ++j) {
    const int kv0 = j * 32;
    v8f sc[2];
#pragma unroll
    for (int c = 0; c < 2; ++c) {
      v8f s4 = v8f_zero();
      const bf16* krow = Kbase + (size_t)(kv0 + c * 16 + ll) * 128 + lh * 8;
#pragma unroll
      for (int kk = 0; kk < 4; ++kk) {
        Frag bk;
        bk.q[0] = *(const uintx4*)(krow + kk * 32);
        bk.q[1] = *(const uintx4*)(krow + kk * 32 + 16);
        s4 = __builtin_amdgcn_wmma_f32_16x16x32_bf16(
            false, qf[kk].v, false, bk.v, (short)0, s4, false, false);
      }
      sc[c] = s4;
    }
    // streaming softmax update for the 32-wide block
#pragma unroll
    for (int r = 0; r < 8; ++r) {
      const int qi = qt * 16 + r + 8 * lh;
      const int kvA = kv0 + ll, kvB = kv0 + 16 + ll;
      float sA = sc[0][r] * scaling + ((kvA > qi) ? -1.0e9f : 0.0f);
      float sB = sc[1][r] * scaling + ((kvB > qi) ? -1.0e9f : 0.0f);
      float t = fmaxf(sA, sB);
#pragma unroll
      for (int msk = 8; msk >= 1; msk >>= 1) t = fmaxf(t, __shfl_xor(t, msk, 16));
      const float mnew = fmaxf(mrow[r], t);
      const float pA = expf(sA - mnew);
      const float pB = expf(sB - mnew);
      float ps = pA + pB;
#pragma unroll
      for (int msk = 8; msk >= 1; msk >>= 1) ps += __shfl_xor(ps, msk, 16);
      const float alpha = expf(mrow[r] - mnew);
      lrow[r] = lrow[r] * alpha + ps;
      mrow[r] = mnew;
#pragma unroll
      for (int t8 = 0; t8 < 8; ++t8) acc[t8][r] = acc[t8][r] * alpha;
      myp[(r + 8 * lh) * 32 + ll]      = (bf16)pA;
      myp[(r + 8 * lh) * 32 + 16 + ll] = (bf16)pB;
    }
    // per-wave LDS bounce: DS ops are in-order within a wave, no barrier.
    Frag pf;
    const bf16* prow = myp + ll * 32 + lh * 8;
    pf.q[0] = *(const uintx4*)(prow);
    pf.q[1] = *(const uintx4*)(prow + 16);
#pragma unroll
    for (int t = 0; t < 8; ++t) {
      const bf16* vrow = Vbase + (size_t)(t * 16 + ll) * 1024 + kv0 + lh * 8;
      Frag bv;
      bv.q[0] = *(const uintx4*)(vrow);
        bv.q[1] = *(const uintx4*)(vrow + 16);
      acc[t] = __builtin_amdgcn_wmma_f32_16x16x32_bf16(
          false, pf.v, false, bv.v, (short)0, acc[t], false, false);
    }
  }

#pragma unroll
  for (int t = 0; t < 8; ++t)
#pragma unroll
    for (int r = 0; r < 8; ++r) {
      const int s = qt * 16 + r + 8 * lh;
      AO[((size_t)b * 1024 + s) * 4096 + h * 128 + t * 16 + ll] =
          acc[t][r] / lrow[r];
    }
}

// ---------------------------------------------------------------------------
// Orchestration. Inputs: hidden, cos, sin, attention_mask(unused: causal mask
// computed inline), Wq, Wk, Wv, Wo. Output: f32 (B,S,H).
// ---------------------------------------------------------------------------
extern "C" void kernel_launch(void* const* d_in, const int* in_sizes, int n_in,
                              void* d_out, int out_size, void* d_ws, size_t ws_size,
                              hipStream_t stream)
{
  (void)in_sizes; (void)n_in; (void)out_size; (void)ws_size;
  const float* hidden = (const float*)d_in[0];
  const float* cosb   = (const float*)d_in[1];
  const float* sinb   = (const float*)d_in[2];
  const float* Wq     = (const float*)d_in[4];
  const float* Wk     = (const float*)d_in[5];
  const float* Wv     = (const float*)d_in[6];
  const float* Wo     = (const float*)d_in[7];

  char* ws = (char*)d_ws;
  size_t off = 0;
  auto alloc = [&](size_t bytes) -> void* {
    void* p = ws + off;
    off += (bytes + 255) & ~(size_t)255;
    return p;
  };

  const long long nX  = 2LL * 1024 * 4096;   // 8,388,608 (B*S*H)
  const long long nWq = 4096LL * 4096;
  const long long nWk = 1024LL * 4096;

  bf16*  Xq  = (bf16*)alloc((size_t)nX  * 2);
  bf16*  Wqq = (bf16*)alloc((size_t)nWq * 2);
  bf16*  Wkq = (bf16*)alloc((size_t)nWk * 2);
  bf16*  Wvq = (bf16*)alloc((size_t)nWk * 2);
  bf16*  Woq = (bf16*)alloc((size_t)nWq * 2);
  float* Qf  = (float*)alloc((size_t)2048 * 4096 * 4);
  float* Kf  = (float*)alloc((size_t)2048 * 1024 * 4);
  float* Vf  = (float*)alloc((size_t)2048 * 1024 * 4);
  bf16*  Qr  = (bf16*)alloc((size_t)2 * 32 * 1024 * 128 * 2);
  bf16*  Kr  = (bf16*)alloc((size_t)2 * 8 * 1024 * 128 * 2);
  bf16*  Vt  = (bf16*)alloc((size_t)2 * 8 * 128 * 1024 * 2);
  float* AO  = (float*)alloc((size_t)2048 * 4096 * 4);
  bf16*  AOq = (bf16*)alloc((size_t)nX * 2);

  dim3 t256(256);

  // MX quantization (activation: mbits=3/max=7.5; weight: mbits=1/max=6.0)
  mx_quant_kernel<<<dim3((unsigned)(nX  / 256)), t256, 0, stream>>>(hidden, Xq,  nX,  3.0f, 7.5f);
  mx_quant_kernel<<<dim3((unsigned)(nWq / 256)), t256, 0, stream>>>(Wq, Wqq, nWq, 1.0f, 6.0f);
  mx_quant_kernel<<<dim3((unsigned)(nWk / 256)), t256, 0, stream>>>(Wk, Wkq, nWk, 1.0f, 6.0f);
  mx_quant_kernel<<<dim3((unsigned)(nWk / 256)), t256, 0, stream>>>(Wv, Wvq, nWk, 1.0f, 6.0f);
  mx_quant_kernel<<<dim3((unsigned)(nWq / 256)), t256, 0, stream>>>(Wo, Woq, nWq, 1.0f, 6.0f);

  // QKV projections (bf16 WMMA)
  gemm_bf16_nt<<<dim3(4096 / 256, 2048 / 128), t256, 0, stream>>>(Xq, Wqq, Qf, 2048, 4096, 4096);
  gemm_bf16_nt<<<dim3(1024 / 256, 2048 / 128), t256, 0, stream>>>(Xq, Wkq, Kf, 2048, 1024, 4096);
  gemm_bf16_nt<<<dim3(1024 / 256, 2048 / 128), t256, 0, stream>>>(Xq, Wvq, Vf, 2048, 1024, 4096);

  // RoPE + layout repack
  rope_pack_kernel<<<dim3((2 * 32 * 1024 * 128) / 256), t256, 0, stream>>>(Qf, cosb, sinb, Qr, 32);
  rope_pack_kernel<<<dim3((2 * 8 * 1024 * 128) / 256),  t256, 0, stream>>>(Kf, cosb, sinb, Kr, 8);
  v_transpose_kernel<<<dim3((2 * 8 * 128 * 1024) / 256), t256, 0, stream>>>(Vf, Vt);

  // Attention: 4096 Q tiles, 8 waves/block
  attn_fwd_kernel<<<dim3(4096 / 8), t256, 0, stream>>>(Qr, Kr, Vt, AO);

  // Output projection (re-quantize activations, bf16 WMMA into d_out)
  mx_quant_kernel<<<dim3((unsigned)(nX / 256)), t256, 0, stream>>>(AO, AOq, nX, 3.0f, 7.5f);
  gemm_bf16_nt<<<dim3(4096 / 256, 2048 / 128), t256, 0, stream>>>(AOq, Woq, (float*)d_out, 2048, 4096, 4096);
}
